// ProbAttention_60644938220067
// MI455X (gfx1250) — matrix operand order repaired and checked by reading, not compiled, not run on hardware
//
#include <hip/hip_runtime.h>
#include <hip/hip_bf16.h>

// ProbSparse attention (Informer-style) for MI455X / gfx1250.
// B=8,H=8,L=2048,D=64, NSEL=40. Bandwidth-bound (~155MB @ 23.3TB/s);
// dense GEMMs use v_wmma_f32_16x16x32_bf16 (probe-confirmed builtin).

#define BH    64     // B*H
#define LEN   2048
#define DIM   64
#define NSEL  40
#define NTIL  128    // LEN/16
#define NTPW  4      // N-tiles per wave in scores kernel
#define KCH   4      // K-split chunks in AV kernel
#define CHN   64     // cumsum chunks
#define CHL   32     // rows per chunk (LEN/CHN)

typedef __attribute__((ext_vector_type(16))) __bf16 v16bf;
typedef __attribute__((ext_vector_type(8)))  float  v8f;

__device__ __forceinline__ v8f wmma_bf16(v16bf a, v16bf b, v8f c) {
    // D = A(16x32 bf16) x B(32x16 bf16) + C(16x16 f32)
    return __builtin_amdgcn_wmma_f32_16x16x32_bf16(
        /*neg_a=*/false, a, /*neg_b=*/false, b,
        /*c_mod=*/(short)0, c, /*reuse_a=*/false, /*reuse_b=*/false);
}

// ---------------------------------------------------------------------------
// 1) measure[b,h,i] = max_j(q_i . k_{rand[i,j]}) - sum_j(...)/LEN   (40 samples)
// ---------------------------------------------------------------------------
__global__ __launch_bounds__(256)
void measure_kernel(const float* __restrict__ q, const float* __restrict__ k,
                    const int* __restrict__ ridx, float* __restrict__ measure) {
    int g = blockIdx.x * blockDim.x + threadIdx.x;   // bh*LEN + i
    if (g >= BH * LEN) return;
    int bh = g >> 11;
    int i  = g & (LEN - 1);
    const float4* qr = (const float4*)(q + (size_t)g * DIM);
    float4 qv[16];
#pragma unroll
    for (int c = 0; c < 16; ++c) qv[c] = qr[c];
    const float* kbase = k + (size_t)bh * LEN * DIM;
    float mx = -__builtin_inff(), sm = 0.0f;
    for (int j = 0; j < NSEL; ++j) {
        int r = ridx[i * NSEL + j];
        if (j + 1 < NSEL) {                               // hide gather latency
            int rn = ridx[i * NSEL + j + 1];
            __builtin_prefetch(kbase + (size_t)rn * DIM, 0, 0);
        }
        const float4* kr = (const float4*)(kbase + (size_t)r * DIM);
        float acc = 0.0f;
#pragma unroll
        for (int c = 0; c < 16; ++c) {
            float4 kv = kr[c];
            acc += qv[c].x * kv.x + qv[c].y * kv.y + qv[c].z * kv.z + qv[c].w * kv.w;
        }
        mx = fmaxf(mx, acc);
        sm += acc;
    }
    measure[g] = mx - sm * (1.0f / (float)LEN);
}

// ---------------------------------------------------------------------------
// 2) top-40 of measure per (b,h); iterative argmax, ties -> lowest index
//    (matches jax.lax.top_k descending order)
// ---------------------------------------------------------------------------
__global__ __launch_bounds__(256)
void topk40(const float* __restrict__ measure, int* __restrict__ idxout) {
    int bh = blockIdx.x;
    __shared__ float vals[LEN];
    __shared__ float rv[256];
    __shared__ int   ri[256];
    int t = threadIdx.x;
    const float* m = measure + (size_t)bh * LEN;
    for (int j = t; j < LEN; j += 256) vals[j] = m[j];
    __syncthreads();
    for (int it = 0; it < NSEL; ++it) {
        float bv = -__builtin_inff(); int bi = LEN;
        for (int j = t; j < LEN; j += 256) {
            float v = vals[j];
            if (v > bv || (v == bv && j < bi)) { bv = v; bi = j; }
        }
        rv[t] = bv; ri[t] = bi;
        __syncthreads();
        for (int s = 128; s > 0; s >>= 1) {
            if (t < s) {
                float ov = rv[t + s]; int oi = ri[t + s];
                if (ov > rv[t] || (ov == rv[t] && oi < ri[t])) { rv[t] = ov; ri[t] = oi; }
            }
            __syncthreads();
        }
        if (t == 0) { idxout[bh * NSEL + it] = ri[0]; vals[ri[0]] = -__builtin_inff(); }
        __syncthreads();
    }
}

// ---------------------------------------------------------------------------
// 3) scores: attn[bh,s,j] = (q[idx_s] . k[j]) / 8, masked j>idx_s -> -inf
//    One wave per (bh, 4 x 16-col tiles): A fragments loaded once, reused
//    across 4 N-tiles -> 24 WMMAs per wave. bf16 fragments per ISA 7.12.2.
// ---------------------------------------------------------------------------
__global__ __launch_bounds__(32)
void scores_wmma(const float* __restrict__ q, const float* __restrict__ k,
                 const int* __restrict__ idxbuf, float* __restrict__ attn) {
    int ntbase = blockIdx.x * NTPW;  // 0,4,...,124
    int bh     = blockIdx.y;         // 0..63
    int lane   = threadIdx.x;        // wave32
    int m      = lane & 15;
    int half   = lane >> 4;
    int kk     = half * 8;
    const float* qbase = q + (size_t)bh * LEN * DIM;
    const float* kbase = k + (size_t)bh * LEN * DIM;

    // A fragments: q_sel tiles (M = selected query, K = d) -- loaded once
    v16bf afrag[3][2];
#pragma unroll
    for (int mt = 0; mt < 3; ++mt) {
        int s = mt * 16 + m; if (s > NSEL - 1) s = NSEL - 1;   // pad clamp
        int row = idxbuf[bh * NSEL + s];
        const float* qr = qbase + (size_t)row * DIM;
#pragma unroll
        for (int kb = 0; kb < 2; ++kb) {
            int K0 = kb * 32 + kk;
            v16bf a;
#pragma unroll
            for (int t = 0; t < 8; ++t) a[t]     = (__bf16)qr[K0 + t];
#pragma unroll
            for (int t = 0; t < 8; ++t) a[8 + t] = (__bf16)qr[K0 + 16 + t];
            afrag[mt][kb] = a;
        }
    }
    const float scale = 0.125f;   // 64^-0.5
    for (int nt = 0; nt < NTPW; ++nt) {
        int ntile = ntbase + nt;
        // B fragments: k^T tile (K = d, N = key j); per lane 2x8 d's (table order)
        int col = ntile * 16 + m;
        const float* kr = kbase + (size_t)col * DIM;
        v16bf bfrag[2];
#pragma unroll
        for (int kb = 0; kb < 2; ++kb) {
            int K0 = kb * 32 + kk;
            v16bf b;
#pragma unroll
            for (int t = 0; t < 8; ++t) b[t]     = (__bf16)kr[K0 + t];
#pragma unroll
            for (int t = 0; t < 8; ++t) b[8 + t] = (__bf16)kr[K0 + 16 + t];
            bfrag[kb] = b;
        }
        v8f acc[3] = {};
#pragma unroll
        for (int mt = 0; mt < 3; ++mt) {
            acc[mt] = wmma_bf16(afrag[mt][0], bfrag[0], acc[mt]);
            acc[mt] = wmma_bf16(afrag[mt][1], bfrag[1], acc[mt]);
        }
        // Store scaled + causal-masked scores (pre-softmax) into attn output
#pragma unroll
        for (int mt = 0; mt < 3; ++mt) {
#pragma unroll
            for (int r = 0; r < 8; ++r) {
                int s = mt * 16 + (half ? (8 + r) : r);
                if (s < NSEL) {
                    int qrow = idxbuf[bh * NSEL + s];
                    int j = ntile * 16 + m;
                    float v = acc[mt][r] * scale;
                    if (j > qrow) v = -__builtin_inff();
                    attn[((size_t)(bh * NSEL + s)) * LEN + j] = v;
                }
            }
        }
    }
}

// ---------------------------------------------------------------------------
// 4) softmax over each attn row (2560 rows x 2048), in place; bf16 copy to ws
// ---------------------------------------------------------------------------
__global__ __launch_bounds__(256)
void softmax_rows(float* __restrict__ attn, unsigned short* __restrict__ attnbf_raw) {
    int row = blockIdx.x;                     // 0..BH*NSEL-1
    float*  p  = attn + (size_t)row * LEN;
    __bf16* pb = reinterpret_cast<__bf16*>(attnbf_raw) + (size_t)row * LEN;
    __shared__ float red[256];
    int t = threadIdx.x;
    float mx = -__builtin_inff();
#pragma unroll
    for (int c = 0; c < 8; ++c) mx = fmaxf(mx, p[t + c * 256]);
    red[t] = mx; __syncthreads();
    for (int s = 128; s > 0; s >>= 1) { if (t < s) red[t] = fmaxf(red[t], red[t + s]); __syncthreads(); }
    mx = red[0]; __syncthreads();
    float e[8]; float sm = 0.0f;
#pragma unroll
    for (int c = 0; c < 8; ++c) { float v = __expf(p[t + c * 256] - mx); e[c] = v; sm += v; }
    red[t] = sm; __syncthreads();
    for (int s = 128; s > 0; s >>= 1) { if (t < s) red[t] += red[t + s]; __syncthreads(); }
    float inv = 1.0f / red[0];
#pragma unroll
    for (int c = 0; c < 8; ++c) {
        float v = e[c] * inv;
        p[t + c * 256]  = v;
        pb[t + c * 256] = (__bf16)v;
    }
}

// ---------------------------------------------------------------------------
// 5) out_sel partials: attn(40x2048) @ v(2048x64), K split across KCH blocks
//    per (b,h) for parallelism. Deterministic: partials summed in fixed order
//    by scatter_sel (no float atomics).
// ---------------------------------------------------------------------------
__global__ __launch_bounds__(128)
void av_wmma(const float* __restrict__ v, const unsigned short* __restrict__ attnbf_raw,
             float* __restrict__ partial) {
    int kc   = blockIdx.x;            // 0..KCH-1
    int bh   = blockIdx.y;            // 0..63
    int w    = threadIdx.x >> 5;      // d-tile 0..3
    int lane = threadIdx.x & 31;
    int m    = lane & 15;
    int half = lane >> 4;
    int kk   = half * 8;
    const float*  vbase = v + (size_t)bh * LEN * DIM;
    const __bf16* ab = reinterpret_cast<const __bf16*>(attnbf_raw) + (size_t)bh * NSEL * LEN;
    int arow[3];
#pragma unroll
    for (int mt = 0; mt < 3; ++mt) { int s = mt * 16 + m; arow[mt] = (s > NSEL - 1) ? NSEL - 1 : s; }
    int d = w * 16 + m;
    v8f acc[3] = {};
    const int KSTEPS = LEN / 32 / KCH;                 // 16 K-steps per chunk
    for (int kt = kc * KSTEPS; kt < (kc + 1) * KSTEPS; ++kt) {
        int K = kt * 32;
        v16bf bb;                                      // B: v[j][d], strided (L2-resident)
#pragma unroll
        for (int t = 0; t < 16; ++t)
            bb[t] = (__bf16)vbase[(size_t)(K + (t < 8 ? kk + t : 16 + kk + (t - 8))) * DIM + d];
#pragma unroll
        for (int mt = 0; mt < 3; ++mt) {
            const __bf16* ar = ab + (size_t)arow[mt] * LEN + K;
            v16bf aa;
#pragma unroll
            for (int t = 0; t < 8; ++t) aa[t]     = ar[kk + t];
#pragma unroll
            for (int t = 0; t < 8; ++t) aa[8 + t] = ar[16 + kk + t];
            acc[mt] = wmma_bf16(aa, bb, acc[mt]);
        }
    }
    float* pout = partial + (size_t)kc * BH * NSEL * DIM;
#pragma unroll
    for (int mt = 0; mt < 3; ++mt) {
#pragma unroll
        for (int r = 0; r < 8; ++r) {
            int s = mt * 16 + (half ? (8 + r) : r);
            if (s < NSEL) pout[((size_t)(bh * NSEL + s)) * DIM + d] = acc[mt][r];
        }
    }
}

// ---------------------------------------------------------------------------
// 6) cumsum(v, axis=L): chunk totals -> exclusive prefix -> final write
// ---------------------------------------------------------------------------
__global__ __launch_bounds__(64)
void chunk_totals(const float* __restrict__ v, float* __restrict__ csum) {
    int ch = blockIdx.x, bh = blockIdx.y, d = threadIdx.x;
    const float* vb = v + ((size_t)bh * LEN + ch * CHL) * DIM + d;
    float acc = 0.0f;
#pragma unroll 4
    for (int r = 0; r < CHL; ++r) acc += vb[(size_t)r * DIM];
    csum[((size_t)bh * CHN + ch) * DIM + d] = acc;
}

__global__ __launch_bounds__(64)
void chunk_prefix(float* __restrict__ csum) {
    int bh = blockIdx.x, d = threadIdx.x;
    float run = 0.0f;
    for (int ch = 0; ch < CHN; ++ch) {
        size_t o = ((size_t)bh * CHN + ch) * DIM + d;
        float t = csum[o]; csum[o] = run; run += t;
    }
}

__global__ __launch_bounds__(64)
void cumsum_write(const float* __restrict__ v, const float* __restrict__ csum,
                  float* __restrict__ out) {
    int ch = blockIdx.x, bh = blockIdx.y, d = threadIdx.x;
    float acc = csum[((size_t)bh * CHN + ch) * DIM + d];
    size_t base = ((size_t)bh * LEN + ch * CHL) * DIM + d;
#pragma unroll 4
    for (int r = 0; r < CHL; ++r) { acc += v[base + (size_t)r * DIM]; out[base + (size_t)r * DIM] = acc; }
}

// ---------------------------------------------------------------------------
// 7) scatter: out[idx_s] = sum_{kc} partial[kc][s]  (fixed order => bitwise
//    deterministic; top-k indices are distinct so no write conflicts)
// ---------------------------------------------------------------------------
__global__ __launch_bounds__(256)
void scatter_sel(const float* __restrict__ partial, const int* __restrict__ idxbuf,
                 float* __restrict__ out) {
    int bh = blockIdx.x;
    const size_t PSTRIDE = (size_t)BH * NSEL * DIM;
    for (int e = threadIdx.x; e < NSEL * DIM; e += 256) {
        int s = e >> 6, d = e & 63;
        size_t o = ((size_t)(bh * NSEL + s)) * DIM + d;
        float acc = partial[o];
#pragma unroll
        for (int kc = 1; kc < KCH; ++kc) acc += partial[(size_t)kc * PSTRIDE + o];
        int row = idxbuf[bh * NSEL + s];
        out[((size_t)bh * LEN + row) * DIM + d] = acc;
    }
}

// ---------------------------------------------------------------------------
extern "C" void kernel_launch(void* const* d_in, const int* in_sizes, int n_in,
                              void* d_out, int out_size, void* d_ws, size_t ws_size,
                              hipStream_t stream) {
    const float* q    = (const float*)d_in[0];
    const float* k    = (const float*)d_in[1];
    const float* v    = (const float*)d_in[2];
    const int*   ridx = (const int*)d_in[3];

    float* out  = (float*)d_out;                     // (B,H,L,D)
    float* attn = out + (size_t)BH * LEN * DIM;      // (B,H,40,L)

    // workspace carve-up (~14.7 MB)
    float* measure = (float*)d_ws;                               // BH*LEN
    int*   idxbuf  = (int*)(measure + (size_t)BH * LEN);         // BH*NSEL
    float* partial = (float*)(idxbuf + (size_t)BH * NSEL);       // KCH*BH*NSEL*DIM
    float* csum    = partial + (size_t)KCH * BH * NSEL * DIM;    // BH*CHN*DIM
    unsigned short* attnbf = (unsigned short*)(csum + (size_t)BH * CHN * DIM); // BH*NSEL*LEN bf16

    measure_kernel<<<(BH * LEN) / 256, 256, 0, stream>>>(q, k, ridx, measure);
    topk40<<<BH, 256, 0, stream>>>(measure, idxbuf);
    scores_wmma<<<dim3(NTIL / NTPW, BH), 32, 0, stream>>>(q, k, idxbuf, attn);
    softmax_rows<<<BH * NSEL, 256, 0, stream>>>(attn, attnbf);
    av_wmma<<<dim3(KCH, BH), 128, 0, stream>>>(v, attnbf, partial);
    chunk_totals<<<dim3(CHN, BH), 64, 0, stream>>>(v, csum);
    chunk_prefix<<<BH, 64, 0, stream>>>(csum);
    cumsum_write<<<dim3(CHN, BH), 64, 0, stream>>>(v, csum, out);
    scatter_sel<<<BH, 256, 0, stream>>>(partial, idxbuf, out);
}
